// GQA_57397942944036
// MI455X (gfx1250) — compile-verified, matrix operand order checked by reading, not run on hardware
//
#include <hip/hip_runtime.h>
#include <hip/hip_fp16.h>

// ---------------------------------------------------------------------------
// GQA attention block on gfx1250 (CDNA5), all matrix math via v_wmma_f32_16x16x32_f16.
// Round 3: attention key-chunks widened to 64 (32 WMMAs per iteration) to
// amortize softmax VALU overhead; exponentials done in exp2/log2 domain.
// ---------------------------------------------------------------------------

#define D_HEAD  128
#define H_Q     16
#define H_K     4
#define D_MODEL 2048
#define BB      2
#define TT      2048

typedef __attribute__((ext_vector_type(16))) _Float16 v16h;
typedef __attribute__((ext_vector_type(8)))  _Float16 v8h;
typedef __attribute__((ext_vector_type(8)))  float    v8f;

__device__ __forceinline__ v8f wmma_f16(v16h a, v16h b, v8f c) {
    // (neg_a, A, neg_b, B, c_mod, C, reuse_a, reuse_b)
    return __builtin_amdgcn_wmma_f32_16x16x32_f16(false, a, false, b, (short)0, c, false, false);
}

// Load a 16x32 (MxK) A-fragment or 32x16 (KxN) B-fragment for an NT GEMM where
// the operand is row-major [rows, ld] with K contiguous.  CDNA5 16-bit WMMA
// layout: lane l holds row (l&15); K chunk base kc = (l>>4)*8, elements
// [kc, kc+8) in the low 8 halves and [kc+16, kc+24) in the high 8 halves.
__device__ __forceinline__ v16h load_frag_nt(const _Float16* __restrict__ base,
                                             int ld, int row0, int k0) {
    const int lane = threadIdx.x & 31;
    const _Float16* p = base + (size_t)(row0 + (lane & 15)) * ld + (k0 + ((lane >> 4) << 3));
    v8h lo = *(const v8h*)(p);
    v8h hi = *(const v8h*)(p + 16);
    v16h r;
#pragma unroll
    for (int i = 0; i < 8; ++i) { r[i] = lo[i]; r[i + 8] = hi[i]; }
    return r;
}

// ---------------------------------------------------------------------------
// fp32 -> fp16 conversion
// ---------------------------------------------------------------------------
__global__ void __launch_bounds__(256)
cvt_f32_to_f16(const float* __restrict__ in, _Float16* __restrict__ out, int n) {
    int i = (blockIdx.x * 256 + threadIdx.x) * 4;
    if (i + 3 < n) {
        float4 v = *(const float4*)(in + i);
        out[i + 0] = (_Float16)v.x;
        out[i + 1] = (_Float16)v.y;
        out[i + 2] = (_Float16)v.z;
        out[i + 3] = (_Float16)v.w;
    }
}

// ---------------------------------------------------------------------------
// NT GEMM: C[M,N] = A[M,K] * B[N,K]^T  (torch Linear layout), f32 accumulate.
// 8 waves / WG; wave tile 32x64 (2x4 WMMA accums); WG tile 128x128.
// Double-buffered fragments: next K-slab loads issue before current WMMAs so
// the matrix pipe overlaps L2 latency (partial s_wait_loadcnt, verified).
// OUT_MODE: 0 = f32 row-major, 1 = f16 row-major, 2 = f16 transposed
//           (Vt[b][n][t], row stride ldt = TT) for the V projection.
// ---------------------------------------------------------------------------
template <int OUT_MODE>
__global__ void __launch_bounds__(256)
gemm_nt_wmma(const _Float16* __restrict__ A, const _Float16* __restrict__ B,
             void* __restrict__ Cout, int M, int N, int K, int ldt) {
    const int wave  = threadIdx.x >> 5;
    const int mBase = blockIdx.y * 128 + (wave >> 1) * 32;
    const int nBase = blockIdx.x * 128 + (wave & 1)  * 64;
    const int lane  = threadIdx.x & 31;

    v8f c[2][4];
#pragma unroll
    for (int i = 0; i < 2; ++i)
#pragma unroll
        for (int j = 0; j < 4; ++j) c[i][j] = v8f{};

    v16h aCur[2], bCur[4], aNxt[2], bNxt[4];
#pragma unroll
    for (int i = 0; i < 2; ++i) aCur[i] = load_frag_nt(A, K, mBase + i * 16, 0);
#pragma unroll
    for (int j = 0; j < 4; ++j) bCur[j] = load_frag_nt(B, K, nBase + j * 16, 0);

    for (int k0 = 0; k0 < K; k0 += 32) {
        const int kn = k0 + 32;
        if (kn < K) {   // issue next K-slab BEFORE consuming the current one
#pragma unroll
            for (int i = 0; i < 2; ++i) aNxt[i] = load_frag_nt(A, K, mBase + i * 16, kn);
#pragma unroll
            for (int j = 0; j < 4; ++j) bNxt[j] = load_frag_nt(B, K, nBase + j * 16, kn);
        }
#pragma unroll
        for (int i = 0; i < 2; ++i)
#pragma unroll
            for (int j = 0; j < 4; ++j)
                c[i][j] = wmma_f16(aCur[i], bCur[j], c[i][j]);
#pragma unroll
        for (int i = 0; i < 2; ++i) aCur[i] = aNxt[i];
#pragma unroll
        for (int j = 0; j < 4; ++j) bCur[j] = bNxt[j];
    }

    // C/D layout: lane l -> n = l&15, rows m = (l>>4)*8 + r for vgpr r.
    const int n    = lane & 15;
    const int mOff = (lane >> 4) * 8;
#pragma unroll
    for (int i = 0; i < 2; ++i)
#pragma unroll
        for (int j = 0; j < 4; ++j)
#pragma unroll
            for (int r = 0; r < 8; ++r) {
                const int m  = mBase + i * 16 + mOff + r;
                const int nn = nBase + j * 16 + n;
                if (OUT_MODE == 0) {
                    ((float*)Cout)[(size_t)m * N + nn] = c[i][j][r];
                } else if (OUT_MODE == 1) {
                    ((_Float16*)Cout)[(size_t)m * N + nn] = (_Float16)c[i][j][r];
                } else {  // transposed f16: rows = feature n, cols = token t (per batch)
                    const int bb = m / TT, t = m % TT;
                    ((_Float16*)Cout)[((size_t)bb * N + nn) * ldt + t] = (_Float16)c[i][j][r];
                }
            }
}

// ---------------------------------------------------------------------------
// Causal flash attention, one wave per (b, hq, 16-query tile), 64-key chunks.
// Q f16 rows [B*T, H_Q*128]; K f16 rows [B*T, H_K*128];
// Vt f16 transposed [B][H_K*128][T]  -> P*V B-fragments are coalesced b128s.
// ctx output rows [B*T, H_Q*128] (f16), fed to the O-projection GEMM.
// Softmax runs in log2 domain (exp2 = raw v_exp_f32).
// ---------------------------------------------------------------------------
__global__ void __launch_bounds__(32)
gqa_flash_attn(const _Float16* __restrict__ Q, const _Float16* __restrict__ K,
               const _Float16* __restrict__ Vt, _Float16* __restrict__ ctx) {
    __shared__ _Float16 ldsP[16 * 64];  // P tile staged C-layout -> A-layout

    const int bh    = blockIdx.x;           // 0 .. B*H_Q-1
    const int b     = bh / H_Q;
    const int hq    = bh % H_Q;
    const int hk    = hq % H_K;             // reference reshape: hq = g*H_K + hk
    const int qBase = blockIdx.y * 16;
    const int lane  = threadIdx.x;

    const int ldq = H_Q * D_HEAD;           // 2048
    const int ldk = H_K * D_HEAD;           // 512
    const _Float16* Qb  = Q  + (size_t)(b * TT) * ldq + hq * D_HEAD;
    const _Float16* Kb  = K  + (size_t)(b * TT) * ldk + hk * D_HEAD;
    const _Float16* Vtb = Vt + ((size_t)b * ldk + hk * D_HEAD) * TT;   // rows = d, cols = t

    // Preload the 16x128 query tile as 4 A-fragments (d chunks of 32).
    v16h qf[4];
#pragma unroll
    for (int d = 0; d < 4; ++d) qf[d] = load_frag_nt(Qb, ldq, qBase, d * 32);

    // Online-softmax state (log2 domain): lane owns rows m = (lane>>4)*8 + r.
    float rowM[8], rowL[8];
    v8f   o[8] = {};                        // 8 d-tiles of 16 -> full D=128
#pragma unroll
    for (int r = 0; r < 8; ++r) { rowM[r] = -__builtin_inff(); rowL[r] = 0.0f; }

    // scores scaled straight into log2 domain: s * (1/sqrt(128)) * log2(e)
    const float scale = 0.08838834764831845f * 1.4426950408889634f;
    const int   mOff  = (lane >> 4) * 8;
    const int   kcol  = lane & 15;
    const int   kEnd  = qBase + 16;             // causal bound (exclusive)

    for (int kb = 0; kb < kEnd; kb += 64) {
        // ---- S = Q K^T for keys [kb, kb+64): 16 WMMAs over d ----
        v8f s[4] = {};
#pragma unroll
        for (int d = 0; d < 4; ++d) {
#pragma unroll
            for (int j = 0; j < 4; ++j) {
                v16h kf = load_frag_nt(Kb, ldk, kb + j * 16, d * 32);
                s[j] = wmma_f16(qf[d], kf, s[j]);
            }
        }

        // ---- scale + causal mask + online softmax (exp2 domain) ----
#pragma unroll
        for (int r = 0; r < 8; ++r) {
            const int q = qBase + mOff + r;
            float v[4];
#pragma unroll
            for (int j = 0; j < 4; ++j) {
                v[j] = s[j][r] * scale;
                if (kb + j * 16 + kcol > q) v[j] = -__builtin_inff();
            }

            float mloc = fmaxf(fmaxf(v[0], v[1]), fmaxf(v[2], v[3]));
#pragma unroll
            for (int off = 1; off < 16; off <<= 1)
                mloc = fmaxf(mloc, __shfl_xor(mloc, off, 16));
            const float mnew = fmaxf(rowM[r], mloc);
            const float corr = (rowM[r] == -__builtin_inff()) ? 0.0f : exp2f(rowM[r] - mnew);

            float pj[4], ls = 0.0f;
#pragma unroll
            for (int j = 0; j < 4; ++j) { pj[j] = exp2f(v[j] - mnew); ls += pj[j]; }
#pragma unroll
            for (int off = 1; off < 16; off <<= 1)
                ls += __shfl_xor(ls, off, 16);

            rowL[r] = rowL[r] * corr + ls;
            rowM[r] = mnew;
#pragma unroll
            for (int t = 0; t < 8; ++t) o[t][r] *= corr;   // rescale O rows

            const int row = mOff + r;                      // C-layout -> LDS (row-major 16x64)
#pragma unroll
            for (int j = 0; j < 4; ++j)
                ldsP[row * 64 + j * 16 + kcol] = (_Float16)pj[j];
        }
        __syncthreads();

        // ---- reload P as two 16x32 A-fragments ----
        v16h pf[2];
#pragma unroll
        for (int h = 0; h < 2; ++h) {
            const _Float16* pp = ldsP + (size_t)(lane & 15) * 64 + h * 32 + ((lane >> 4) << 3);
            v8h lo = *(const v8h*)(pp);
            v8h hi = *(const v8h*)(pp + 16);
#pragma unroll
            for (int i = 0; i < 8; ++i) { pf[h][i] = lo[i]; pf[h][i + 8] = hi[i]; }
        }

        // ---- O += P V : 16 WMMAs (8 d-tiles x 2 key halves) ----
#pragma unroll
        for (int t = 0; t < 8; ++t) {
            v16h vf0 = load_frag_nt(Vtb, TT, t * 16, kb);
            v16h vf1 = load_frag_nt(Vtb, TT, t * 16, kb + 32);
            o[t] = wmma_f16(pf[0], vf0, o[t]);
            o[t] = wmma_f16(pf[1], vf1, o[t]);
        }
        __syncthreads();
    }

    // ---- normalize and emit context tile ----
#pragma unroll
    for (int t = 0; t < 8; ++t)
#pragma unroll
        for (int r = 0; r < 8; ++r) {
            size_t row = (size_t)(b * TT + qBase + mOff + r);
            ctx[row * ldq + hq * D_HEAD + t * 16 + kcol] = (_Float16)(o[t][r] / rowL[r]);
        }
}

// ---------------------------------------------------------------------------
// Host-side orchestration
// ---------------------------------------------------------------------------
extern "C" void kernel_launch(void* const* d_in, const int* in_sizes, int n_in,
                              void* d_out, int out_size, void* d_ws, size_t ws_size,
                              hipStream_t stream) {
    const float* x  = (const float*)d_in[0];
    const float* wq = (const float*)d_in[1];
    const float* wk = (const float*)d_in[2];
    const float* wv = (const float*)d_in[3];
    const float* wo = (const float*)d_in[4];

    const size_t M   = (size_t)BB * TT;          // 4096 token rows
    const size_t nX  = M * D_MODEL;              // 8388608
    const size_t nWq = (size_t)D_MODEL * D_MODEL;
    const size_t nWk = (size_t)(H_K * D_HEAD) * D_MODEL;
    const size_t nQ  = M * (H_Q * D_HEAD);
    const size_t nK  = M * (H_K * D_HEAD);

    _Float16* p    = (_Float16*)d_ws;            // ~80 MB of f16 staging
    _Float16* x16  = p;  p += nX;
    _Float16* wq16 = p;  p += nWq;
    _Float16* wk16 = p;  p += nWk;
    _Float16* wv16 = p;  p += nWk;
    _Float16* wo16 = p;  p += nWq;
    _Float16* q16  = p;  p += nQ;
    _Float16* k16  = p;  p += nK;
    _Float16* vt16 = p;  p += nK;                // V stored transposed [B][512][T]
    _Float16* ctx  = p;  p += nQ;

    // 1) fp32 -> fp16 conversions
    cvt_f32_to_f16<<<dim3(nX  / 1024), 256, 0, stream>>>(x,  x16,  (int)nX);
    cvt_f32_to_f16<<<dim3(nWq / 1024), 256, 0, stream>>>(wq, wq16, (int)nWq);
    cvt_f32_to_f16<<<dim3(nWk / 1024), 256, 0, stream>>>(wk, wk16, (int)nWk);
    cvt_f32_to_f16<<<dim3(nWk / 1024), 256, 0, stream>>>(wv, wv16, (int)nWk);
    cvt_f32_to_f16<<<dim3(nWq / 1024), 256, 0, stream>>>(wo, wo16, (int)nWq);

    // 2) projections: Q = x wq^T, K = x wk^T (row-major f16), V transposed.
    gemm_nt_wmma<1><<<dim3(D_MODEL / 128, M / 128), 256, 0, stream>>>(
        x16, wq16, q16, (int)M, D_MODEL, D_MODEL, 0);
    gemm_nt_wmma<1><<<dim3((H_K * D_HEAD) / 128, M / 128), 256, 0, stream>>>(
        x16, wk16, k16, (int)M, H_K * D_HEAD, D_MODEL, 0);
    gemm_nt_wmma<2><<<dim3((H_K * D_HEAD) / 128, M / 128), 256, 0, stream>>>(
        x16, wv16, vt16, (int)M, H_K * D_HEAD, D_MODEL, TT);

    // 3) causal GQA flash attention (64-key chunks)
    gqa_flash_attn<<<dim3(BB * H_Q, TT / 16), 32, 0, stream>>>(q16, k16, vt16, ctx);

    // 4) output projection: out = ctx wo^T (f32 out straight to d_out)
    gemm_nt_wmma<0><<<dim3(D_MODEL / 128, M / 128), 256, 0, stream>>>(
        ctx, wo16, d_out, (int)M, D_MODEL, D_MODEL, 0);
}
